// SignalAttention_53987738911346
// MI455X (gfx1250) — matrix operand order repaired and checked by reading, not compiled, not run on hardware
//
#include <hip/hip_runtime.h>
#include <hip/hip_bf16.h>

// ---------------------------------------------------------------------------
// Types for CDNA5 WMMA / TDM
// ---------------------------------------------------------------------------
typedef __attribute__((ext_vector_type(16))) __bf16       v16bf;
typedef __attribute__((ext_vector_type(8)))  float        v8f;
typedef __attribute__((ext_vector_type(4)))  unsigned int v4u;
typedef __attribute__((ext_vector_type(8)))  unsigned int v8u;

#define B_  4
#define T_  2048
#define C_  256          // IN == H == 256
#define BT_ (B_ * T_)    // 8192 rows
#define WIN 33
#define HALF 16

// workspace layout (float offsets)
#define OFF_Q     0
#define OFF_K     (BT_ * C_)               // 2097152
#define OFF_V     (2 * BT_ * C_)
#define OFF_MEAN  (3 * BT_ * C_)           // 3*4*256 = 3072 floats
#define OFF_RSTD  (OFF_MEAN + 3 * B_ * C_)
#define OFF_PART  (OFF_RSTD + 3 * B_ * C_) // 4*32*256 floats
#define OFF_WFRAG (OFF_PART + B_ * 32 * C_) // 6330368, 32B-aligned
// wfrag: 3 mats * 16 nt * 8 k0c * 32 lanes * (16 hi + 16 lo) bf16 = 393216 bf16

// ---------------------------------------------------------------------------
// Kernel 0: pre-swizzle W into WMMA B-fragment layout, bf16 hi/lo split.
// B[k][n] = W[n][k]; fragment lane L: N = L%16, elem e -> K = k0 + (e/8)*16 + (L/16)*8 + e%8
// grid = 768 x 256 threads, one element each.
// ---------------------------------------------------------------------------
__global__ __launch_bounds__(256)
void wconv_kernel(const float* __restrict__ Wq, const float* __restrict__ Wk,
                  const float* __restrict__ Wv, __bf16* __restrict__ wfrag)
{
    const int id = blockIdx.x * 256 + threadIdx.x;   // 0 .. 196607
    const int e    = id & 15;
    const int lane = (id >> 4) & 31;
    const int k0c  = (id >> 9) & 7;
    const int nt   = (id >> 12) & 15;
    const int mat  = id >> 16;                        // 0..2
    const float* Wm = (mat == 0) ? Wq : (mat == 1) ? Wk : Wv;
    const float v = Wm[(size_t)(nt * 16 + (lane & 15)) * C_ + k0c * 32 + (e >> 3) * 16 + (lane >> 4) * 8 + (e & 7)];
    const __bf16 h = (__bf16)v;
    const __bf16 l = (__bf16)(v - (float)h);
    const size_t base = ((size_t)(id >> 4)) * 32;     // per-(frag,lane) 32 bf16
    wfrag[base + e]      = h;
    wfrag[base + 16 + e] = l;
}

// ---------------------------------------------------------------------------
// Kernel 1: fused QKV GEMM  y = x @ W.T + b  (fp32 in/out, bf16x3 WMMA core)
// grid = 3 * 512 one-wave blocks. Each wave: 16-row M tile x 256 cols, K=256.
// ---------------------------------------------------------------------------
__global__ __launch_bounds__(32)
void gemm_qkv_wmma(const float* __restrict__ x,
                   const float* __restrict__ bq, const float* __restrict__ bk,
                   const float* __restrict__ bv,
                   const __bf16* __restrict__ wfrag,
                   float* __restrict__ ws)
{
    const int gid   = blockIdx.x;
    const int mat   = gid >> 9;          // 0:q 1:k 2:v
    const int mtile = gid & 511;
    const int m0    = mtile * 16;

    const float* bias = (mat == 0) ? bq : (mat == 1) ? bk : bv;
    float*       outp = ws + (size_t)mat * (BT_ * C_);

    const int lane = threadIdx.x;        // 0..31
    const int lo16 = lane & 15;
    const int hi   = lane >> 4;          // 0/1

    const float* xrow = x + (size_t)(m0 + lo16) * C_;

    v8f acc[16];
#pragma unroll
    for (int nt = 0; nt < 16; ++nt)
#pragma unroll
        for (int r = 0; r < 8; ++r) acc[nt][r] = 0.0f;

    for (int k0 = 0; k0 < C_; k0 += 32) {
        const int k0c = k0 >> 5;
        // ---- A fragment: 16x32 bf16, lane L holds M=L%16,
        //      elems 0..7 -> K = k0 + hi*8 + e ; elems 8..15 -> K = k0+16+hi*8+(e-8)
        float av[16];
        {
            const float4 a0 = *reinterpret_cast<const float4*>(xrow + k0 + hi * 8);
            const float4 a1 = *reinterpret_cast<const float4*>(xrow + k0 + hi * 8 + 4);
            const float4 a2 = *reinterpret_cast<const float4*>(xrow + k0 + 16 + hi * 8);
            const float4 a3 = *reinterpret_cast<const float4*>(xrow + k0 + 16 + hi * 8 + 4);
            av[0]=a0.x; av[1]=a0.y; av[2]=a0.z; av[3]=a0.w;
            av[4]=a1.x; av[5]=a1.y; av[6]=a1.z; av[7]=a1.w;
            av[8]=a2.x; av[9]=a2.y; av[10]=a2.z; av[11]=a2.w;
            av[12]=a3.x; av[13]=a3.y; av[14]=a3.z; av[15]=a3.w;
        }
        v16bf ah, al;
#pragma unroll
        for (int e = 0; e < 16; ++e) {
            __bf16 h = (__bf16)av[e];
            ah[e] = h;
            al[e] = (__bf16)(av[e] - (float)h);
        }

        for (int nt = 0; nt < 16; ++nt) {
            // ---- B fragments: precomputed bf16 hi/lo, 2 x 32B vector loads
            const __bf16* wf = wfrag +
                ((size_t)(((mat * 16 + nt) * 8 + k0c) * 32 + lane) * 32);
            const v16bf bh = *reinterpret_cast<const v16bf*>(wf);
            const v16bf bl = *reinterpret_cast<const v16bf*>(wf + 16);
            // bf16x3 emulated-fp32 product
            acc[nt] = __builtin_amdgcn_wmma_f32_16x16x32_bf16(
                false, ah, false, bh, (short)0, acc[nt], false, false);
            acc[nt] = __builtin_amdgcn_wmma_f32_16x16x32_bf16(
                false, ah, false, bl, (short)0, acc[nt], false, false);
            acc[nt] = __builtin_amdgcn_wmma_f32_16x16x32_bf16(
                false, al, false, bh, (short)0, acc[nt], false, false);
        }
    }

    // ---- store: C/D layout -> lane L, VGPR r : M = r + 8*(L/16), N = L%16
#pragma unroll
    for (int nt = 0; nt < 16; ++nt) {
        const int  ncol = nt * 16 + lo16;
        const float bb  = bias[ncol];
#pragma unroll
        for (int r = 0; r < 8; ++r) {
            const int mrow = m0 + r + 8 * hi;
            outp[(size_t)mrow * C_ + ncol] = acc[nt][r] + bb;
        }
    }
}

// ---------------------------------------------------------------------------
// Kernel 2: per (tensor, b, h) mean / rstd over T.   grid = 12, block = 1024.
// ---------------------------------------------------------------------------
__global__ __launch_bounds__(1024)
void stats_kernel(const float* __restrict__ ws,
                  float* __restrict__ means, float* __restrict__ rstds)
{
    const int bi   = blockIdx.x;   // 0..11
    const int mat  = bi >> 2;
    const int b    = bi & 3;
    const int h    = threadIdx.x & 255;
    const int part = threadIdx.x >> 8;   // 0..3, 512 t each

    const float* base = ws + (size_t)mat * (BT_ * C_)
                           + ((size_t)(b * T_ + part * 512)) * C_ + h;
    float s = 0.0f, s2 = 0.0f;
    for (int i = 0; i < 512; ++i) {
        float v = base[(size_t)i * C_];
        s += v; s2 += v * v;
    }
    __shared__ float sh_s[1024];
    __shared__ float sh_s2[1024];
    sh_s[threadIdx.x]  = s;
    sh_s2[threadIdx.x] = s2;
    __syncthreads();
    if (part == 0) {
#pragma unroll
        for (int p = 1; p < 4; ++p) {
            s  += sh_s[h + 256 * p];
            s2 += sh_s2[h + 256 * p];
        }
        const float mean = s * (1.0f / T_);
        float var = s2 * (1.0f / T_) - mean * mean;
        var = fmaxf(var, 0.0f);
        means[mat * (B_ * C_) + b * C_ + h] = mean;
        rstds[mat * (B_ * C_) + b * C_ + h] = rsqrtf(var + 1e-5f);
    }
}

// ---------------------------------------------------------------------------
// Kernel 3: banded attention, f32.  grid = B*32 = 128 blocks, 256 threads.
// k/v slab (96 rows x 256) DMA'd into 192KB LDS by the Tensor Data Mover,
// then edge-replicated (index clamp) and instance-normalized in place.
// ---------------------------------------------------------------------------
__global__ __launch_bounds__(256)
void attn_kernel(const float* __restrict__ ws,
                 const float* __restrict__ means, const float* __restrict__ rstds,
                 float* __restrict__ partial)
{
    extern __shared__ float smem[];         // [96*256] kn | [96*256] vn
    float* kl = smem;
    float* vl = smem + 96 * C_;

    const int blk = blockIdx.x;
    const int b   = blk >> 5;
    const int tc  = blk & 31;
    const int t0  = tc * 64;

    const int tid  = threadIdx.x;
    const int lane = tid & 31;
    const int wid  = tid >> 5;

    const float* qraw = ws + OFF_Q;
    const float* kraw = ws + OFF_K;
    const float* vraw = ws + OFF_V;

    // valid global row range for this block's slab (rows t0-16 .. t0+79)
    const int row_lo    = t0 - HALF;
    const int row_start = max(row_lo, 0);
    const int row_end   = min(t0 + 79, T_ - 1);
    const int nrows     = row_end - row_start + 1;      // 96 interior, 80 at edges
    const int miss_top  = row_start - row_lo;           // 0 or 16
    const int last      = miss_top + nrows - 1;         // last filled LDS row

    // ---- TDM: async DMA the raw k/v slab into LDS (one issuing wave)
    if (tid < 32) {
        const unsigned kl_lds = (unsigned)(size_t)kl;
        const unsigned vl_lds = (unsigned)(size_t)vl;
        const unsigned long long ka =
            (unsigned long long)(size_t)(kraw + (size_t)(b * T_ + row_start) * C_);
        const unsigned long long va =
            (unsigned long long)(size_t)(vraw + (size_t)(b * T_ + row_start) * C_);

        v8u g1;                                  // shared dims for both copies
        g1[0] = 2u << 16;                        // data_size = 4 bytes
        g1[1] = (unsigned)C_ << 16;              // tensor_dim0 = 256
        g1[2] = ((unsigned)nrows & 0xffffu) << 16; // tensor_dim1 = nrows
        g1[3] = (unsigned)C_ << 16;              // tile_dim0 = 256
        g1[4] = (unsigned)nrows;                 // tile_dim1 = nrows, tile_dim2 = 0
        g1[5] = (unsigned)C_;                    // tensor_dim0_stride = 256
        g1[6] = 0u;
        g1[7] = 0u;

        v4u g0k;
        g0k[0] = 1u;                                            // count = 1
        g0k[1] = kl_lds + (unsigned)miss_top * (C_ * 4u);       // lds_addr
        g0k[2] = (unsigned)ka;                                  // global_addr lo
        g0k[3] = (unsigned)((ka >> 32) & 0x01FFFFFFull) | (2u << 30); // hi | type=2
        asm volatile("tensor_load_to_lds %0, %1" :: "s"(g0k), "s"(g1) : "memory");

        v4u g0v;
        g0v[0] = 1u;
        g0v[1] = vl_lds + (unsigned)miss_top * (C_ * 4u);
        g0v[2] = (unsigned)va;
        g0v[3] = (unsigned)((va >> 32) & 0x01FFFFFFull) | (2u << 30);
        asm volatile("tensor_load_to_lds %0, %1" :: "s"(g0v), "s"(g1) : "memory");

        __builtin_amdgcn_s_wait_tensorcnt(0);
    }
    __syncthreads();

    // ---- edge replication (reference's index clamp), per column h = tid
    {
        const int h = tid;
        for (int i = 0; i < miss_top; ++i) {
            kl[i * C_ + h] = kl[miss_top * C_ + h];
            vl[i * C_ + h] = vl[miss_top * C_ + h];
        }
        for (int i = last + 1; i < 96; ++i) {
            kl[i * C_ + h] = kl[last * C_ + h];
            vl[i * C_ + h] = vl[last * C_ + h];
        }
    }
    __syncthreads();

    // ---- in-place instance norm of the slab
    {
        const int h = tid;
        const float km = means[B_ * C_     + b * C_ + h];
        const float kr = rstds[B_ * C_     + b * C_ + h];
        const float vm = means[2 * B_ * C_ + b * C_ + h];
        const float vr = rstds[2 * B_ * C_ + b * C_ + h];
        for (int i = 0; i < 96; ++i) {
            kl[i * C_ + h] = (kl[i * C_ + h] - km) * kr;
            vl[i * C_ + h] = (vl[i * C_ + h] - vm) * vr;
        }
    }
    __syncthreads();

    // per-lane q stats for h = lane + 32*j
    float qm[8], qr[8];
#pragma unroll
    for (int j = 0; j < 8; ++j) {
        const int hh = lane + 32 * j;
        qm[j] = means[b * C_ + hh];
        qr[j] = rstds[b * C_ + hh];
    }

    float acc[8];
#pragma unroll
    for (int j = 0; j < 8; ++j) acc[j] = 0.0f;

    for (int tt = 0; tt < 8; ++tt) {
        const int tl = wid * 8 + tt;        // 0..63
        const int t  = t0 + tl;
        const float* qrow = qraw + (size_t)(b * T_ + t) * C_;
        float qv[8];
#pragma unroll
        for (int j = 0; j < 8; ++j)
            qv[j] = (qrow[lane + 32 * j] - qm[j]) * qr[j];

        float s[WIN];
        float smax = -3.0e38f;
#pragma unroll
        for (int w = 0; w < WIN; ++w) {
            const int li = tl + w;          // 0..95
            float p = 0.0f;
#pragma unroll
            for (int j = 0; j < 8; ++j)
                p += qv[j] * kl[li * C_ + lane + 32 * j];
            p += __shfl_xor(p, 16);
            p += __shfl_xor(p, 8);
            p += __shfl_xor(p, 4);
            p += __shfl_xor(p, 2);
            p += __shfl_xor(p, 1);
            const int row = t + w - HALF;
            const float sv = (row >= 0 && row < T_) ? p * 0.0625f : -3.0e38f;
            s[w] = sv;
            smax = fmaxf(smax, sv);
        }

        float psum = 0.0f;
        float o[8];
#pragma unroll
        for (int j = 0; j < 8; ++j) o[j] = 0.0f;
#pragma unroll
        for (int w = 0; w < WIN; ++w) {
            const float pw = __expf(s[w] - smax);
            psum += pw;
            const int li = tl + w;
#pragma unroll
            for (int j = 0; j < 8; ++j)
                o[j] += pw * vl[li * C_ + lane + 32 * j];
        }
        const float inv = 1.0f / psum;
#pragma unroll
        for (int j = 0; j < 8; ++j) acc[j] += o[j] * inv;
    }

    // ---- cross-wave reduction (reuse LDS), then deterministic partial store
    __syncthreads();
#pragma unroll
    for (int j = 0; j < 8; ++j)
        kl[wid * C_ + lane + 32 * j] = acc[j];
    __syncthreads();

    float tot = 0.0f;
#pragma unroll
    for (int w8 = 0; w8 < 8; ++w8) tot += kl[w8 * C_ + tid];
    partial[(size_t)(b * 32 + tc) * C_ + tid] = tot;
}

// ---------------------------------------------------------------------------
// Kernel 4: finalize mean over T.  grid = 4, block = 256.
// ---------------------------------------------------------------------------
__global__ __launch_bounds__(256)
void finalize_kernel(const float* __restrict__ partial, float* __restrict__ out)
{
    const int b = blockIdx.x;
    const int h = threadIdx.x;
    float s = 0.0f;
#pragma unroll
    for (int tc = 0; tc < 32; ++tc)
        s += partial[(size_t)(b * 32 + tc) * C_ + h];
    out[b * C_ + h] = s * (1.0f / T_);
}

// ---------------------------------------------------------------------------
extern "C" void kernel_launch(void* const* d_in, const int* in_sizes, int n_in,
                              void* d_out, int out_size, void* d_ws, size_t ws_size,
                              hipStream_t stream)
{
    (void)in_sizes; (void)n_in; (void)out_size; (void)ws_size;
    const float* x  = (const float*)d_in[0];
    const float* Wq = (const float*)d_in[1];
    const float* bq = (const float*)d_in[2];
    const float* Wk = (const float*)d_in[3];
    const float* bk = (const float*)d_in[4];
    const float* Wv = (const float*)d_in[5];
    const float* bv = (const float*)d_in[6];

    float*  ws      = (float*)d_ws;
    float*  means   = ws + OFF_MEAN;
    float*  rstds   = ws + OFF_RSTD;
    float*  partial = ws + OFF_PART;
    __bf16* wfrag   = (__bf16*)(ws + OFF_WFRAG);

    wconv_kernel<<<dim3(768), dim3(256), 0, stream>>>(Wq, Wk, Wv, wfrag);
    gemm_qkv_wmma<<<dim3(3 * 512), dim3(32), 0, stream>>>(x, bq, bk, bv, wfrag, ws);
    stats_kernel<<<dim3(12), dim3(1024), 0, stream>>>(ws, means, rstds);
    attn_kernel<<<dim3(B_ * 32), dim3(256), 96 * C_ * 2 * sizeof(float), stream>>>(
        ws, means, rstds, partial);
    finalize_kernel<<<dim3(B_), dim3(256), 0, stream>>>(partial, (float*)d_out);
}
